// RGCNNaivePsychologyLinkPredict_8693013807317
// MI455X (gfx1250) — matrix operand order repaired and checked by reading, not compiled, not run on hardware
//
#include <hip/hip_runtime.h>

#define DDIM 768

typedef __bf16 bf16_t;
typedef __attribute__((ext_vector_type(16))) __bf16 v16bf;
typedef __attribute__((ext_vector_type(8)))  __bf16 v8bf;
typedef __attribute__((ext_vector_type(8)))  float   v8f;

__device__ __forceinline__ bf16_t f32_to_bf16(float f) {
  union { float f; unsigned u; } v; v.f = f;
  unsigned r = v.u + 0x7FFFu + ((v.u >> 16) & 1u);   // round-to-nearest-even
  return __builtin_bit_cast(bf16_t, (unsigned short)(r >> 16));
}

// ---------------- elementwise helpers ----------------

__global__ __launch_bounds__(256) void k_zero_f32(float* __restrict__ p, size_t n) {
  size_t i = (size_t)blockIdx.x * blockDim.x + threadIdx.x;
  size_t s = (size_t)gridDim.x * blockDim.x;
  for (; i < n; i += s) p[i] = 0.0f;
}

__global__ __launch_bounds__(256) void k_zero_i32(int* __restrict__ p, int n) {
  int i = blockIdx.x * blockDim.x + threadIdx.x;
  if (i < n) p[i] = 0;
}

__global__ __launch_bounds__(256) void k_cast_bf16(const float* __restrict__ in,
                                                   bf16_t* __restrict__ out, size_t n) {
  size_t i = (size_t)blockIdx.x * blockDim.x + threadIdx.x;
  size_t s = (size_t)gridDim.x * blockDim.x;
  for (; i < n; i += s) out[i] = f32_to_bf16(in[i]);
}

// in: [count][K][N] f32 row-major  ->  out: [count][N][K] bf16 (B transposed)
__global__ __launch_bounds__(256) void k_transpose_cast(const float* __restrict__ in,
                                                        bf16_t* __restrict__ out, int count) {
  const size_t mat = (size_t)DDIM * DDIM;
  size_t total = (size_t)count * mat;
  size_t i = (size_t)blockIdx.x * blockDim.x + threadIdx.x;
  size_t s = (size_t)gridDim.x * blockDim.x;
  for (; i < total; i += s) {
    size_t c = i / mat, rem = i % mat;
    int n = (int)(rem / DDIM), k = (int)(rem % DDIM);
    out[i] = f32_to_bf16(in[c * mat + (size_t)k * DDIM + n]);
  }
}

// h_next = relu(agg + selfW); also emit bf16 copy for next layer's GEMMs
__global__ __launch_bounds__(256) void k_combine(const float* __restrict__ agg,
                                                 const float* __restrict__ selfw,
                                                 float* __restrict__ h,
                                                 bf16_t* __restrict__ hb, size_t n) {
  size_t i = (size_t)blockIdx.x * blockDim.x + threadIdx.x;
  size_t s = (size_t)gridDim.x * blockDim.x;
  for (; i < n; i += s) {
    float v = agg[i] + selfw[i];
    v = v > 0.0f ? v : 0.0f;
    h[i] = v;
    hb[i] = f32_to_bf16(v);
  }
}

// ---------------- WMMA GEMM ----------------
// C[M x Nc] (f32) = A[M x K] (bf16 row-major) * Bt^T, Bt stored [Nc x K] (bf16).
// Block: 128 threads = 4 waves, tile 64 rows x 128 cols; wave tile 32x64 (2x4 WMMA tiles).
// If offs != nullptr, rows are stored only when their global compact index
// (c0 + row) lies in [offs[rsel], offs[rsel+1]) — relation-bucketed masked store.
__global__ __launch_bounds__(128) void k_gemm_bf16(
    const bf16_t* __restrict__ A, const bf16_t* __restrict__ Bt,
    float* __restrict__ C, int M, int Nc, int K,
    const int* __restrict__ offs, int rsel, int c0) {
  const int lane = threadIdx.x & 31;
  const int wave = threadIdx.x >> 5;
  const int half = lane >> 4;        // 0/1: which 16-lane group
  const int l16  = lane & 15;
  const int m0 = blockIdx.x * 64 + (wave & 1) * 32;
  const int n0 = blockIdx.y * 128 + (wave >> 1) * 64;

  int lo_r = 0, hi_r = M;
  if (offs) {
    lo_r = offs[rsel] - c0;
    hi_r = offs[rsel + 1] - c0;
    int b0 = blockIdx.x * 64;
    if (b0 >= hi_r || b0 + 64 <= lo_r) return;   // block disjoint from this relation's range
  }

  // A fragment rows (clamped; out-of-range rows computed but never stored)
  const int ar0 = (m0 + l16)      < M ? (m0 + l16)      : (M - 1);
  const int ar1 = (m0 + 16 + l16) < M ? (m0 + 16 + l16) : (M - 1);
  const bf16_t* __restrict__ Arow0 = A + (size_t)ar0 * K;
  const bf16_t* __restrict__ Arow1 = A + (size_t)ar1 * K;
  // B fragment: lane = column n0+l16; per-lane K window starts at half*16
  const bf16_t* __restrict__ Bcol = Bt + (size_t)(n0 + l16) * K + half * 16;

  v8f acc[2][4];
#pragma unroll
  for (int mt = 0; mt < 2; ++mt)
#pragma unroll
    for (int nt = 0; nt < 4; ++nt)
#pragma unroll
      for (int i = 0; i < 8; ++i) acc[mt][nt][i] = 0.0f;

  for (int k0 = 0; k0 < K; k0 += 32) {
    __builtin_prefetch((const void*)(Arow0 + k0 + 64), 0, 3);
    __builtin_prefetch((const void*)(Arow1 + k0 + 64), 0, 3);
    // A fragments: lane holds row (lane&15); K = (lane>>4)*16 + [0..15]
    v8bf lo0 = *reinterpret_cast<const v8bf*>(Arow0 + k0 + half * 16);
    v8bf hi0 = *reinterpret_cast<const v8bf*>(Arow0 + k0 + half * 16 + 8);
    v8bf lo1 = *reinterpret_cast<const v8bf*>(Arow1 + k0 + half * 16);
    v8bf hi1 = *reinterpret_cast<const v8bf*>(Arow1 + k0 + half * 16 + 8);
    v16bf a[2];
#pragma unroll
    for (int i = 0; i < 8; ++i) {
      a[0][i] = lo0[i]; a[0][i + 8] = hi0[i];
      a[1][i] = lo1[i]; a[1][i + 8] = hi1[i];
    }
    // B fragments: lane holds column (lane&15); 16 contiguous K from (lane>>4)*16
    v16bf b[4];
#pragma unroll
    for (int nt = 0; nt < 4; ++nt)
      b[nt] = *reinterpret_cast<const v16bf*>(Bcol + (size_t)nt * 16 * K + k0);

#pragma unroll
    for (int mt = 0; mt < 2; ++mt)
#pragma unroll
      for (int nt = 0; nt < 4; ++nt)
        acc[mt][nt] = __builtin_amdgcn_wmma_f32_16x16x32_bf16(
            false, a[mt], false, b[nt], (short)0, acc[mt][nt], false, false);
  }

  // C layout: lane holds N = lane&15, VGPR v holds M = v + 8*(lane>>4)
#pragma unroll
  for (int mt = 0; mt < 2; ++mt) {
#pragma unroll
    for (int v = 0; v < 8; ++v) {
      int row = m0 + mt * 16 + half * 8 + v;
      if (row >= M || row < lo_r || row >= hi_r) continue;
      size_t base = (size_t)row * Nc + n0 + l16;
#pragma unroll
      for (int nt = 0; nt < 4; ++nt) C[base + nt * 16] = acc[mt][nt][v];
    }
  }
}

// ---------------- edge aggregation (one wave per edge) ----------------
__global__ __launch_bounds__(256) void k_scatter(
    const float* __restrict__ hW, const float* __restrict__ norm,
    const int* __restrict__ src, const int* __restrict__ dst,
    const int* __restrict__ rel, int rsel, int E, float* __restrict__ agg) {
  int e = blockIdx.x * 8 + (threadIdx.x >> 5);
  if (e >= E) return;
  if (rel[e] != rsel) return;
  const float* __restrict__ srow = hW + (size_t)src[e] * DDIM;
  float* __restrict__ drow = agg + (size_t)dst[e] * DDIM;
  float w = norm[e];
  for (int j = threadIdx.x & 31; j < DDIM; j += 32) atomicAdd(drow + j, w * srow[j]);
}

// ---------------- relation bucketing for scoring ----------------
__global__ __launch_bounds__(256) void k_count(const int* __restrict__ pos_rel,
                                               const int* __restrict__ neg_rel,
                                               int T, int* __restrict__ cnt) {
  int t = blockIdx.x * blockDim.x + threadIdx.x;
  if (t >= 2 * T) return;
  int er = t < T ? pos_rel[t] : neg_rel[t - T];
  atomicAdd(&cnt[er], 1);
}

__global__ void k_scan(const int* __restrict__ cnt, int* __restrict__ offs, int R) {
  if (threadIdx.x == 0 && blockIdx.x == 0) {
    int s = 0;
    for (int r = 0; r < R; ++r) { offs[r] = s; s += cnt[r]; }
    offs[R] = s;
  }
}

__global__ __launch_bounds__(256) void k_fill_idx(const int* __restrict__ pos_rel,
                                                  const int* __restrict__ neg_rel, int T,
                                                  const int* __restrict__ offs,
                                                  int* __restrict__ cur, int* __restrict__ idx) {
  int t = blockIdx.x * blockDim.x + threadIdx.x;
  if (t >= 2 * T) return;
  int er = t < T ? pos_rel[t] : neg_rel[t - T];
  int pos = atomicAdd(&cur[er], 1);
  idx[offs[er] + pos] = t;
}

// gather emb rows for compact targets [c0, c0+cnt) into bf16 chunk
__global__ __launch_bounds__(256) void k_gather(
    const float* __restrict__ h, const int* __restrict__ idx,
    const int* __restrict__ pos_src, const int* __restrict__ neg_src,
    int T, int c0, int cnt, bf16_t* __restrict__ s) {
  int i = blockIdx.x * 8 + (threadIdx.x >> 5);
  if (i >= cnt) return;
  int t = idx[c0 + i];
  int es = t < T ? pos_src[t] : neg_src[t - T];
  const float* __restrict__ row = h + (size_t)es * DDIM;
  bf16_t* __restrict__ srow = s + (size_t)i * DDIM;
  for (int j = threadIdx.x & 31; j < DDIM; j += 32) srow[j] = f32_to_bf16(row[j]);
}

// score[t] = dot(sW[i], emb[eo(t)])  — wave-per-row with shuffle reduction
__global__ __launch_bounds__(256) void k_dot(
    const float* __restrict__ sW, const float* __restrict__ h,
    const int* __restrict__ idx, const int* __restrict__ pos_dst,
    const int* __restrict__ neg_dst, int T, int c0, int cnt, float* __restrict__ out) {
  int i = blockIdx.x * 8 + (threadIdx.x >> 5);
  if (i >= cnt) return;
  int t = idx[c0 + i];
  int eo = t < T ? pos_dst[t] : neg_dst[t - T];
  const float* __restrict__ o = h + (size_t)eo * DDIM;
  const float* __restrict__ srow = sW + (size_t)i * DDIM;
  float acc = 0.0f;
  for (int j = threadIdx.x & 31; j < DDIM; j += 32) acc = fmaf(srow[j], o[j], acc);
#pragma unroll
  for (int off = 16; off > 0; off >>= 1) acc += __shfl_xor(acc, off, 32);
  if ((threadIdx.x & 31) == 0) out[t] = acc;
}

// ---------------- host orchestration ----------------

extern "C" void kernel_launch(void* const* d_in, const int* in_sizes, int n_in,
                              void* d_out, int out_size, void* d_ws, size_t ws_size,
                              hipStream_t stream) {
  (void)n_in; (void)out_size; (void)ws_size;
  const int d = DDIM;
  const float* node_feat  = (const float*)d_in[0];
  const float* edge_norm  = (const float*)d_in[1];
  const float* W_rel      = (const float*)d_in[2];
  const float* W_self     = (const float*)d_in[3];
  const float* w_relation = (const float*)d_in[4];
  const int* src     = (const int*)d_in[5];
  const int* dst     = (const int*)d_in[6];
  const int* rel     = (const int*)d_in[7];
  const int* pos_src = (const int*)d_in[8];
  const int* pos_rel = (const int*)d_in[9];
  const int* pos_dst = (const int*)d_in[10];
  const int* neg_src = (const int*)d_in[11];
  const int* neg_rel = (const int*)d_in[12];
  const int* neg_dst = (const int*)d_in[13];

  const int N = in_sizes[0] / d;
  const int E = in_sizes[1];
  const int L = in_sizes[3] / (d * d);
  const int R = in_sizes[4] / (d * d);
  const int T = in_sizes[8];
  const int TT = 2 * T;
  const int CH = 8192;

  char* p = (char*)d_ws;
  auto take = [&p](size_t bytes) {
    void* q = (void*)p;
    p += (bytes + 255) & ~(size_t)255;
    return q;
  };
  const size_t Nd = (size_t)N * d;
  float*  h_f32    = (float*) take(Nd * 4);
  float*  agg      = (float*) take(Nd * 4);
  float*  tmp      = (float*) take(Nd * 4);
  bf16_t* h_bf     = (bf16_t*)take(Nd * 2);
  bf16_t* Wrel_bt  = (bf16_t*)take((size_t)L * R * d * d * 2);
  bf16_t* Wself_bt = (bf16_t*)take((size_t)L * d * d * 2);
  bf16_t* wrel_bt  = (bf16_t*)take((size_t)R * d * d * 2);
  bf16_t* s_bf     = (bf16_t*)take((size_t)CH * d * 2);
  float*  sW       = (float*) take((size_t)CH * d * 4);
  int* cnts = (int*)take((size_t)(2 * R) * 4);   // [0..R) counts, [R..2R) cursors
  int* offs = (int*)take((size_t)(R + 1) * 4);
  int* idx  = (int*)take((size_t)TT * 4);

  // one-time weight transpose + cast to bf16 column-major
  k_transpose_cast<<<2048, 256, 0, stream>>>(W_rel, Wrel_bt, L * R);
  k_transpose_cast<<<512,  256, 0, stream>>>(W_self, Wself_bt, L);
  k_transpose_cast<<<1024, 256, 0, stream>>>(w_relation, wrel_bt, R);
  k_cast_bf16<<<4096, 256, 0, stream>>>(node_feat, h_bf, Nd);

  dim3 gemm_grid((unsigned)((N + 63) / 64), (unsigned)(d / 128));
  for (int l = 0; l < L; ++l) {
    k_zero_f32<<<4096, 256, 0, stream>>>(agg, Nd);
    for (int r = 0; r < R; ++r) {
      k_gemm_bf16<<<gemm_grid, 128, 0, stream>>>(
          h_bf, Wrel_bt + (size_t)(l * R + r) * d * d, tmp, N, d, d, nullptr, 0, 0);
      k_scatter<<<(E + 7) / 8, 256, 0, stream>>>(tmp, edge_norm, src, dst, rel, r, E, agg);
    }
    k_gemm_bf16<<<gemm_grid, 128, 0, stream>>>(
        h_bf, Wself_bt + (size_t)l * d * d, tmp, N, d, d, nullptr, 0, 0);
    k_combine<<<4096, 256, 0, stream>>>(agg, tmp, h_f32, h_bf, Nd);
  }

  // bucket targets by relation (avoids the reference's 8x masked-GEMM waste)
  k_zero_i32<<<1, 256, 0, stream>>>(cnts, 2 * R);
  k_count<<<(TT + 255) / 256, 256, 0, stream>>>(pos_rel, neg_rel, T, cnts);
  k_scan<<<1, 32, 0, stream>>>(cnts, offs, R);
  k_fill_idx<<<(TT + 255) / 256, 256, 0, stream>>>(pos_rel, neg_rel, T, offs, cnts + R, idx);

  for (int c0 = 0; c0 < TT; c0 += CH) {
    int cnt = (TT - c0) < CH ? (TT - c0) : CH;
    k_gather<<<(cnt + 7) / 8, 256, 0, stream>>>(h_f32, idx, pos_src, neg_src, T, c0, cnt, s_bf);
    dim3 sg((unsigned)((cnt + 63) / 64), (unsigned)(d / 128));
    for (int r = 0; r < R; ++r)
      k_gemm_bf16<<<sg, 128, 0, stream>>>(s_bf, wrel_bt + (size_t)r * d * d,
                                          sW, cnt, d, d, offs, r, c0);
    k_dot<<<(cnt + 7) / 8, 256, 0, stream>>>(sW, h_f32, idx, pos_dst, neg_dst, T, c0, cnt,
                                             (float*)d_out);
  }
}
